// BaselineCCNN_87033217286323
// MI455X (gfx1250) — matrix operand order repaired and checked by reading, not compiled, not run on hardware
//
// CCNN forward for MI455X (gfx1250), wave32 + WMMA f16 (f32 accumulate) + TDM.
//
// Pipeline (all on `stream`):
//   k_maxtype  -> BOS type id
//   k_setup    -> times/dt/types (with BOS) + f16 embedding gather
//   k_prep_*   -> convert & swizzle all weights into WMMA-fragment-contiguous f16
//   per layer: k_mlp   (temporal enc + 3 WMMA GEMMs -> h3, f16)
//              k_cconv (TDM tensor_load_to_lds stages the (rows x taps x 128) h3
//                       tile; rank-5 outer-product pre-sum S; one WMMA GEMM
//                       against W4|b4|Ws|bs folded matrix; leaky epilogue)
//   k_conv_head (discrete conv as WMMA GEMM + head WMMA GEMM + softplus/loglik)
#include <hip/hip_runtime.h>
#include <cstdint>
#include <math.h>

typedef _Float16 h16;
typedef __attribute__((ext_vector_type(16))) _Float16 v16h;
typedef __attribute__((ext_vector_type(8)))  float    v8f;

#if __has_builtin(__builtin_amdgcn_tensor_load_to_lds) && __has_builtin(__builtin_amdgcn_s_wait_tensorcnt)
#define CCNN_HAS_TDM 1
typedef unsigned int v4u __attribute__((ext_vector_type(4)));
typedef int          v4i __attribute__((ext_vector_type(4)));
typedef int          v8i __attribute__((ext_vector_type(8)));
#endif

#define BS        4
#define LSEQ      1024
#define LP        1025
#define ROWS      (BS*LP)       // 4100
#define ROWS_PAD  4112
#define TILES     257           // ceil(4100/16)
#define MROWS     (BS*5*LP)     // 20500 (b,k,l) MLP rows
#define MROWS_PAD 20512
#define MTILES    1282
#define KSTEPS_BMAT 264         // (132 slots * 64 c) / 32
#define SLOPE     0.1f

static __device__ __forceinline__ float leaky(float x){ return x >= 0.f ? x : SLOPE*x; }

static __device__ __forceinline__ v8f wmma_f16(v16h a, v16h b, v8f c){
  return __builtin_amdgcn_wmma_f32_16x16x32_f16(false, a, false, b, (short)0, c, false, false);
}

// A fragment (16x32 f16) from an LDS tile laid out [16 rows][KD halfs].
// Contiguous K order used here (B prep matches): lanes 0-15 -> M=lane, K={0..15}; lanes 16-31 -> M=lane-16, K={16..31}.
static __device__ __forceinline__ v16h afrag_lds(const h16* base, int KD, int K0, int lane){
  int row = lane & 15;
  int kb  = (lane & 16) ? 16 : 0;
  const uint32_t* p0 = (const uint32_t*)(base + row*KD + K0 + kb);
  const uint32_t* p1 = (const uint32_t*)(base + row*KD + K0 + kb + 8);
  union { v16h v; uint32_t u[8]; } r;
#pragma unroll
  for (int j = 0; j < 4; ++j) { r.u[j] = p0[j]; r.u[4+j] = p1[j]; }
  return r.v;
}

// B fragment (32x16 f16) from pre-swizzled global weights: 16 contiguous halfs/lane.
static __device__ __forceinline__ v16h bfrag_g(const h16* g, int frag, int lane){
  return *(const v16h*)(g + ((size_t)frag*32 + lane)*16);
}

// ---------------------------------------------------------------- k_maxtype
__global__ __launch_bounds__(256) void k_maxtype(const int* __restrict__ ty, int n, int* __restrict__ bos){
  __shared__ int red[256];
  int t = threadIdx.x, m = 0;
  for (int i = t; i < n; i += 256) m = max(m, ty[i]);
  red[t] = m; __syncthreads();
  for (int s = 128; s > 0; s >>= 1) { if (t < s) red[t] = max(red[t], red[t+s]); __syncthreads(); }
  if (t == 0) bos[0] = red[0] + 1;
}

// ---------------------------------------------------------------- k_setup
__global__ __launch_bounds__(256) void k_setup(const float* __restrict__ et, const int* __restrict__ ety,
                        const float* __restrict__ emb, const int* __restrict__ bosp,
                        float* __restrict__ times, float* __restrict__ dts,
                        int* __restrict__ types, h16* __restrict__ enc0){
  int row = blockIdx.x*4 + (threadIdx.x >> 6);
  int c   = threadIdx.x & 63;
  if (row >= ROWS_PAD) return;
  int b = row / LP, l = row - b*LP;
  float tv = 0.f, dv = 0.f, ev = 0.f; int ty = 0;
  if (row < ROWS) {
    if (l == 0) { ty = bosp[0]; }
    else {
      tv = et[b*LSEQ + (l-1)];
      ty = ety[b*LSEQ + (l-1)];
      dv = tv - ((l >= 2) ? et[b*LSEQ + (l-2)] : 0.f);
    }
    ev = emb[ty*64 + c];
  }
  if (c == 0) { times[row] = tv; dts[row] = dv; types[row] = ty; }
  enc0[(size_t)row*64 + c] = (h16)ev;
}

// ------------------------------------------------- weight swizzle kernels
// frag layout in memory: [n_tile][kstep][lane(32)][e(16)] halfs; value = src[K][N] with
// K = kstep*32 + (lane<16 ? 0:16) + e, N = n_tile*16 + (lane&15).
__global__ void k_prep_lin(const float* __restrict__ src, h16* __restrict__ dst, int Kdim, int N){
  int ksteps = Kdim >> 5, ntiles = N >> 4;
  int total = ntiles*ksteps*512;
  for (int idx = blockIdx.x*blockDim.x + threadIdx.x; idx < total; idx += gridDim.x*blockDim.x){
    int e = idx & 15, lane = (idx >> 4) & 31, f = idx >> 9;
    int ks = f % ksteps, nt = f / ksteps;
    int kk = ks*32 + ((lane & 16) ? 16 : 0) + e;
    int col = nt*16 + (lane & 15);
    dst[idx] = (h16)src[kk*N + col];
  }
}

// Folded cont-conv matrix: K-slots h=0..127 -> W4[h,c,d]; 128 -> b4[c,d] (hits masked-f sum);
// 129 -> Ws[c,d] (skip, hits raw feats); 130 -> bs[d] at c==0 (hits const 1); 131 -> zero pad.
__global__ void k_prep_bmat(const float* __restrict__ W4, const float* __restrict__ b4,
                            const float* __restrict__ Ws, const float* __restrict__ bs,
                            h16* __restrict__ dst){
  int total = 4*KSTEPS_BMAT*512;
  for (int idx = blockIdx.x*blockDim.x + threadIdx.x; idx < total; idx += gridDim.x*blockDim.x){
    int e = idx & 15, lane = (idx >> 4) & 31, f = idx >> 9;
    int ks = f % KSTEPS_BMAT;
    int kk = ks*32 + ((lane & 16) ? 16 : 0) + e;
    int col = (f / KSTEPS_BMAT)*16 + (lane & 15);
    int h = kk >> 6, c = kk & 63;
    float v = 0.f;
    if      (h < 128)  v = W4[h*4096 + c*64 + col];
    else if (h == 128) v = b4[c*64 + col];
    else if (h == 129) v = Ws[c*64 + col];
    else if (h == 130) v = (c == 0) ? bs[col] : 0.f;
    dst[idx] = (h16)v;
  }
}

// im2col conv weights: K-index kk = m*64+c, value = dconv_w[o][c][m]
__global__ void k_prep_conv(const float* __restrict__ w, h16* __restrict__ dst){
  const int ksteps = 10;
  int total = 4*ksteps*512;
  for (int idx = blockIdx.x*blockDim.x + threadIdx.x; idx < total; idx += gridDim.x*blockDim.x){
    int e = idx & 15, lane = (idx >> 4) & 31, f = idx >> 9;
    int ks = f % ksteps, nt = f / ksteps;
    int kk = ks*32 + ((lane & 16) ? 16 : 0) + e;
    int col = nt*16 + (lane & 15);
    int m = kk >> 6, c = kk & 63;
    dst[idx] = (h16)w[col*320 + c*5 + m];
  }
}

// head: cols 0..49 = type head, col 50 = lambda head, rest zero
__global__ void k_prep_head(const float* __restrict__ tw, const float* __restrict__ lw,
                            h16* __restrict__ dst){
  const int ksteps = 2;
  int total = 4*ksteps*512;
  for (int idx = blockIdx.x*blockDim.x + threadIdx.x; idx < total; idx += gridDim.x*blockDim.x){
    int e = idx & 15, lane = (idx >> 4) & 31, f = idx >> 9;
    int ks = f % ksteps, nt = f / ksteps;
    int kk = ks*32 + ((lane & 16) ? 16 : 0) + e;   // c index 0..63
    int col = nt*16 + (lane & 15);
    float v = 0.f;
    if (col < 50)       v = tw[kk*50 + col];
    else if (col == 50) v = lw[kk];
    dst[idx] = (h16)v;
  }
}

// ---------------------------------------------------------------- k_mlp
__global__ __launch_bounds__(128) void k_mlp(
    const float* __restrict__ times, const int* __restrict__ types,
    const h16* __restrict__ W1s, const float* __restrict__ b1,
    const h16* __restrict__ W2s, const float* __restrict__ b2,
    const h16* __restrict__ W3s, const float* __restrict__ b3,
    h16* __restrict__ h3, int dil){
  __shared__ h16 X0[16*64];
  __shared__ h16 H1[16*128];
  __shared__ h16 H2[16*128];
  const int tid = threadIdx.x, lane = tid & 31, wave = tid >> 5;
  const int tile = blockIdx.x;

  { // sinusoidal encoding of masked delta-times
    int r  = tid >> 3;
    int i0 = (tid & 7)*8;
    int rm = tile*16 + r;
    float dtv = 0.f;
    if (rm < MROWS) {
      int b = rm / (5*LP);
      int rem = rm - b*(5*LP);
      int k = rem / LP, l = rem - k*LP;
      int off = (4 - k)*dil;
      int rc = b*LP + l;
      if (l >= off && types[rc] != 0 && types[rc-off] != 0)
        dtv = times[rc] - times[rc-off];
    }
#pragma unroll
    for (int j = 0; j < 8; ++j) {
      int i = i0 + j;
      float rr = dtv * __expf(-(float)(i >> 1) * (9.210340371976184f/32.f)); // 10000^(-(i/2)/32)
      X0[r*64 + i] = (h16)((i & 1) ? __cosf(rr) : __sinf(rr));
    }
  }
  __syncthreads();

  const int colb = lane & 15, rb = (lane & 16) ? 8 : 0;
  { // H1 = leaky(X0 @ W1 + b1)   (16x64)@(64x128)
    v16h a0 = afrag_lds(X0, 64, 0, lane);
    v16h a1 = afrag_lds(X0, 64, 32, lane);
#pragma unroll
    for (int t = 0; t < 2; ++t) {
      int nt = wave + 4*t;
      v8f acc = {0.f,0.f,0.f,0.f,0.f,0.f,0.f,0.f};
      acc = wmma_f16(a0, bfrag_g(W1s, nt*2+0, lane), acc);
      acc = wmma_f16(a1, bfrag_g(W1s, nt*2+1, lane), acc);
      int col = nt*16 + colb; float bb = b1[col];
#pragma unroll
      for (int i = 0; i < 8; ++i) H1[(rb+i)*128 + col] = (h16)leaky(acc[i] + bb);
    }
  }
  __syncthreads();
  { // H2 = leaky(H1 @ W2 + b2)   (16x128)@(128x128)
#pragma unroll
    for (int t = 0; t < 2; ++t) {
      int nt = wave + 4*t;
      v8f acc = {0.f,0.f,0.f,0.f,0.f,0.f,0.f,0.f};
#pragma unroll
      for (int ks = 0; ks < 4; ++ks)
        acc = wmma_f16(afrag_lds(H1, 128, ks*32, lane), bfrag_g(W2s, nt*4+ks, lane), acc);
      int col = nt*16 + colb; float bb = b2[col];
#pragma unroll
      for (int i = 0; i < 8; ++i) H2[(rb+i)*128 + col] = (h16)leaky(acc[i] + bb);
    }
  }
  __syncthreads();
  { // h3 = leaky(H2 @ W3 + b3) -> global
#pragma unroll
    for (int t = 0; t < 2; ++t) {
      int nt = wave + 4*t;
      v8f acc = {0.f,0.f,0.f,0.f,0.f,0.f,0.f,0.f};
#pragma unroll
      for (int ks = 0; ks < 4; ++ks)
        acc = wmma_f16(afrag_lds(H2, 128, ks*32, lane), bfrag_g(W3s, nt*4+ks, lane), acc);
      int col = nt*16 + colb; float bb = b3[col];
#pragma unroll
      for (int i = 0; i < 8; ++i) {
        int rm = tile*16 + rb + i;
        if (rm < MROWS) h3[(size_t)rm*128 + col] = (h16)leaky(acc[i] + bb);
      }
    }
  }
}

// ---------------------------------------------------------------- k_cconv
__global__ __launch_bounds__(128) void k_cconv(
    const h16* __restrict__ encPrev, const int* __restrict__ types,
    const h16* __restrict__ Bmat, const h16* __restrict__ h3,
    h16* __restrict__ encNext, int dil){
  __shared__ h16 h3t[16*5*128];   // 20 KB  [row][k][h]
  __shared__ h16 ftl[16*5*64];    // 10 KB, mask-zeroed lagged features
  __shared__ h16 Sc[16*16*64];    // 32 KB, one 16-slot chunk of S
  const int tid = threadIdx.x, lane = tid & 31, wave = tid >> 5;
  const int tile = blockIdx.x;
  const int row0 = tile*16;
  const int b0 = row0 / LP, l0 = row0 - b0*LP;
  const int avail = LP - l0;      // rows of batch b0 in this tile (zero-fill beyond)

#ifdef CCNN_HAS_TDM
  // TDM: DMA the 3D tile h3[(b0*5+k)*LP + l0 + y][x] (x=128 halfs, y=16 rows, z=5 taps)
  // into LDS. tensor_dim1 = avail makes the TDM zero-fill rows past the end of the
  // batch-b0 sequence (the OOB-read-returns-zero rule), which is exactly our padding.
  if (wave == 0) {
    uint64_t ga   = (uint64_t)(uintptr_t)(h3 + (size_t)(b0*5*LP + l0)*128);
    uint32_t lds  = (uint32_t)(uintptr_t)h3t;   // LDS byte offset (low 32 bits of flat)
    uint32_t dim1 = (uint32_t)avail;
    v4u g0;
    g0[0] = 1u;                                              // count=1 valid user D#
    g0[1] = lds;                                             // lds_addr
    g0[2] = (uint32_t)(ga & 0xffffffffu);                    // global_addr[31:0]
    g0[3] = (uint32_t)((ga >> 32) & 0x01ffffffu) | (2u<<30); // global_addr[56:32], type=2
    const uint32_t td0 = 128, tile0 = 128, tile1 = 16, tile2 = 5;
    const uint64_t s0 = 128;                 // y stride (elements)
    const uint64_t s1 = (uint64_t)LP * 128;  // z (tap) stride (elements)
    v8i g1;
    g1[0] = (int)(1u << 16);                                          // data_size = 2B
    g1[1] = (int)((td0 & 0xffffu) << 16);                             // tensor_dim0 lo16
    g1[2] = (int)((td0 >> 16) | ((dim1 & 0xffffu) << 16));            // dim0 hi | dim1 lo
    g1[3] = (int)((dim1 >> 16) | (tile0 << 16));                      // dim1 hi | tile_dim0
    g1[4] = (int)(tile1 | (tile2 << 16));                             // tile_dim1 | tile_dim2
    g1[5] = (int)(uint32_t)(s0 & 0xffffffffu);                        // dim0_stride lo32
    g1[6] = (int)(uint32_t)((uint32_t)(s0 >> 32) | (((uint32_t)s1 & 0xffffu) << 16)); // s0 hi16 | s1 lo16
    g1[7] = (int)(uint32_t)(s1 >> 16);                                // s1[47:16]
    v4i g2; g2[0] = 5; g2[1] = 1; g2[2] = 0; g2[3] = 0;               // tensor_dim2=5
    v4i g3; g3[0] = 0; g3[1] = 0; g3[2] = 0; g3[3] = 0;
#if defined(__clang_major__) && __clang_major__ >= 23
    v8i gz = {0,0,0,0,0,0,0,0};
    __builtin_amdgcn_tensor_load_to_lds(g0, g1, g2, g3, gz, 0);
#else
    __builtin_amdgcn_tensor_load_to_lds(g0, g1, g2, g3, 0);
#endif
    __builtin_amdgcn_s_wait_tensorcnt(0);
  }
  __syncthreads();
  // fix-up rows past the batch-b0 boundary that belong to the next batch
  if (avail < 16) {
    for (int idx = tid; idx < 16*5*64; idx += 128) {
      int r = idx / 320, rem = idx - r*320, k = rem >> 6, hd = rem & 63;
      if (r < avail) continue;          // staged (or correctly zero-filled) by TDM
      int row = row0 + r;
      uint32_t v = 0;
      if (row < ROWS) {
        int b = row / LP, l = row - b*LP;
        v = ((const uint32_t*)h3)[(size_t)((b*5 + k)*LP + l)*64 + hd];
      }
      ((uint32_t*)h3t)[idx] = v;
    }
  }
#else
  { // manual staging fallback
    const uint32_t* s32 = (const uint32_t*)h3;
    uint32_t* d32 = (uint32_t*)h3t;
    for (int idx = tid; idx < 16*5*64; idx += 128) {
      int r = idx / 320, rem = idx - r*320, k = rem >> 6, hd = rem & 63;
      int row = row0 + r;
      uint32_t v = 0;
      if (row < ROWS) {
        int b = row / LP, l = row - b*LP;
        v = s32[(size_t)((b*5 + k)*LP + l)*64 + hd];
      }
      d32[idx] = v;
    }
  }
#endif
  for (int idx = tid; idx < 16*5*64; idx += 128) { // masked lag_f
    int r = idx / 320, rem = idx - r*320, k = rem >> 6, c = rem & 63;
    int row = row0 + r;
    h16 v = (h16)0.f;
    if (row < ROWS) {
      int l = row % LP;
      int off = (4 - k)*dil;
      if (l >= off && types[row] != 0 && types[row-off] != 0)
        v = encPrev[(size_t)(row-off)*64 + c];
    }
    ftl[idx] = v;
  }
  __syncthreads();

  v8f acc = {0.f,0.f,0.f,0.f,0.f,0.f,0.f,0.f};
  for (int hc = 0; hc < 8; ++hc) {
    { // S[r][hh][c] = sum_k h3[r,k,h] * f[r,k,c]  (rank-5 outer-product pre-sum)
      int r = tid >> 3, c0 = (tid & 7)*8;
      float fr[5][8];
#pragma unroll
      for (int k = 0; k < 5; ++k)
#pragma unroll
        for (int j = 0; j < 8; ++j) fr[k][j] = (float)ftl[(r*5+k)*64 + c0 + j];
      for (int hh = 0; hh < 16; ++hh) {
        int h = hc*16 + hh;
        float hv[5];
#pragma unroll
        for (int k = 0; k < 5; ++k) hv[k] = (float)h3t[(r*5+k)*128 + h];
#pragma unroll
        for (int j = 0; j < 8; ++j) {
          float s = 0.f;
#pragma unroll
          for (int k = 0; k < 5; ++k) s += hv[k]*fr[k][j];
          Sc[(r*16+hh)*64 + c0 + j] = (h16)s;
        }
      }
    }
    __syncthreads();
#pragma unroll 4
    for (int s = 0; s < 32; ++s) {
      int ks = hc*32 + s;
      __builtin_prefetch(Bmat + ((size_t)(wave*KSTEPS_BMAT + ks + 4)*32 + lane)*16, 0, 1);
      acc = wmma_f16(afrag_lds(Sc, 1024, s*32, lane),
                     bfrag_g(Bmat, wave*KSTEPS_BMAT + ks, lane), acc);
    }
    __syncthreads();
  }
  // tail slots: 128=masked-f sum (b4), 129=raw feats (skip Ws), 130=const1 (bs), 131=pad
  for (int idx = tid; idx < 16*256; idx += 128) {
    int r = idx >> 8, q = idx & 255, slot = q >> 6, c = q & 63;
    int row = row0 + r;
    float v = 0.f;
    if (slot == 0) {
#pragma unroll
      for (int k = 0; k < 5; ++k) v += (float)ftl[(r*5+k)*64 + c];
    } else if (slot == 1) {
      if (row < ROWS) v = (float)encPrev[(size_t)row*64 + c];
    } else if (slot == 2) {
      v = (c == 0) ? 1.f : 0.f;
    }
    Sc[idx] = (h16)v;
  }
  __syncthreads();
#pragma unroll
  for (int s = 0; s < 8; ++s)
    acc = wmma_f16(afrag_lds(Sc, 256, s*32, lane),
                   bfrag_g(Bmat, wave*KSTEPS_BMAT + 256 + s, lane), acc);

  int col = wave*16 + (lane & 15), rb = (lane & 16) ? 8 : 0;
#pragma unroll
  for (int i = 0; i < 8; ++i) {
    int row = row0 + rb + i;
    if (row < ROWS_PAD)
      encNext[(size_t)row*64 + col] = (h16)((row < ROWS) ? leaky(acc[i]) : 0.f);
  }
}

// ---------------------------------------------------------------- k_conv_head
__global__ __launch_bounds__(128) void k_conv_head(
    const h16* __restrict__ enc2, const h16* __restrict__ Bconv,
    const float* __restrict__ dconv_b, const h16* __restrict__ Bhead,
    const float* __restrict__ type_b, const float* __restrict__ lam_b,
    const float* __restrict__ dts, float* __restrict__ outp){
  __shared__ h16 At[16*320];
  __shared__ h16 Ht[16*64];
  const int tid = threadIdx.x, lane = tid & 31, wave = tid >> 5;
  const int tile = blockIdx.x;

  for (int idx = tid; idx < 16*320; idx += 128) { // im2col, zero-padded per batch row
    int r = idx / 320, q = idx - r*320, m = q >> 6, c = q & 63;
    int row = tile*16 + r;
    h16 v = (h16)0.f;
    if (row < ROWS) {
      int b = row / LP, l = row - b*LP;
      int ls = l + m - 2;
      if (ls >= 0 && ls < LP) v = enc2[(size_t)(b*LP + ls)*64 + c];
    }
    At[idx] = v;
  }
  __syncthreads();

  const int colb = lane & 15, rb = (lane & 16) ? 8 : 0;
  v8f acc = {0.f,0.f,0.f,0.f,0.f,0.f,0.f,0.f};
#pragma unroll
  for (int s = 0; s < 10; ++s)
    acc = wmma_f16(afrag_lds(At, 320, s*32, lane), bfrag_g(Bconv, wave*10+s, lane), acc);
  int col = wave*16 + colb;
  float bb = dconv_b[col];
#pragma unroll
  for (int i = 0; i < 8; ++i) Ht[(rb+i)*64 + col] = (h16)leaky(acc[i] + bb);
  __syncthreads();

  v8f acc2 = {0.f,0.f,0.f,0.f,0.f,0.f,0.f,0.f};
#pragma unroll
  for (int s = 0; s < 2; ++s)
    acc2 = wmma_f16(afrag_lds(Ht, 64, s*32, lane), bfrag_g(Bhead, wave*2+s, lane), acc2);
#pragma unroll
  for (int i = 0; i < 8; ++i) {
    int row = tile*16 + rb + i;
    if (row >= ROWS) continue;
    if (col < 50) {
      outp[(size_t)row*50 + col] = acc2[i] + type_b[col];
    } else if (col == 50) {
      float z = acc2[i] + lam_b[0];
      float lam = (z > 20.f) ? z : log1pf(__expf(z));        // softplus
      outp[(size_t)ROWS*50 + row] = logf(lam + 1e-8f) - lam * dts[row];
    }
  }
}

// ---------------------------------------------------------------- launch
extern "C" void kernel_launch(void* const* d_in, const int* in_sizes, int n_in,
                              void* d_out, int out_size, void* d_ws, size_t ws_size,
                              hipStream_t stream) {
  (void)in_sizes; (void)n_in; (void)out_size;
  const float* event_times = (const float*)d_in[0];
  const int*   event_types = (const int*)d_in[1];
  const float* emb         = (const float*)d_in[2];
  const float* cp[20];
  for (int i = 0; i < 20; ++i) cp[i] = (const float*)d_in[3+i];   // 2 layers x (W1,b1,W2,b2,W3,b3,W4,b4,Ws,bs)
  const float* dconv_w = (const float*)d_in[23];
  const float* dconv_b = (const float*)d_in[24];
  const float* type_w  = (const float*)d_in[25];
  const float* type_b  = (const float*)d_in[26];
  const float* lam_w   = (const float*)d_in[27];
  const float* lam_b   = (const float*)d_in[28];

  char* p = (char*)d_ws;
  size_t used = 0;
  auto carve = [&](size_t bytes) -> char* {
    char* r = p + used;
    used += (bytes + 255) & ~(size_t)255;
    return r;
  };
  int*   bosp  = (int*)  carve(256);
  float* times = (float*)carve((size_t)ROWS_PAD*4);
  float* dts   = (float*)carve((size_t)ROWS_PAD*4);
  int*   types = (int*)  carve((size_t)ROWS_PAD*4);
  h16*   encA  = (h16*)  carve((size_t)ROWS_PAD*64*2);
  h16*   encB  = (h16*)  carve((size_t)ROWS_PAD*64*2);
  h16*   h3    = (h16*)  carve((size_t)MROWS_PAD*128*2);
  h16 *W1s[2], *W2s[2], *W3s[2], *Bm[2];
  for (int l = 0; l < 2; ++l) {
    W1s[l] = (h16*)carve(8192*2);
    W2s[l] = (h16*)carve(16384*2);
    W3s[l] = (h16*)carve(16384*2);
    Bm[l]  = (h16*)carve((size_t)4*KSTEPS_BMAT*512*2);
  }
  h16* Bconv = (h16*)carve(20480*2);
  h16* Bhead = (h16*)carve(4096*2);
  if (used > ws_size) return;   // workspace too small: do nothing (defined behavior)

  k_maxtype<<<1, 256, 0, stream>>>(event_types, BS*LSEQ, bosp);
  k_setup<<<ROWS_PAD/4, 256, 0, stream>>>(event_times, event_types, emb, bosp,
                                          times, dts, types, encA);
  for (int l = 0; l < 2; ++l) {
    const float* const* P = cp + l*10;
    k_prep_lin<<<32, 256, 0, stream>>>(P[0], W1s[l], 64, 128);
    k_prep_lin<<<64, 256, 0, stream>>>(P[2], W2s[l], 128, 128);
    k_prep_lin<<<64, 256, 0, stream>>>(P[4], W3s[l], 128, 128);
    k_prep_bmat<<<512, 256, 0, stream>>>(P[6], P[7], P[8], P[9], Bm[l]);
  }
  k_prep_conv<<<32, 256, 0, stream>>>(dconv_w, Bconv);
  k_prep_head<<<8, 256, 0, stream>>>(type_w, lam_w, Bhead);

  h16* eIn = encA; h16* eOut = encB;
  const int dils[2] = {1, 2};
  for (int l = 0; l < 2; ++l) {
    const float* const* P = cp + l*10;
    k_mlp<<<MTILES, 128, 0, stream>>>(times, types, W1s[l], P[1], W2s[l], P[3],
                                      W3s[l], P[5], h3, dils[l]);
    k_cconv<<<TILES, 128, 0, stream>>>(eIn, types, Bm[l], h3, eOut, dils[l]);
    h16* t = eIn; eIn = eOut; eOut = t;
  }
  k_conv_head<<<TILES, 128, 0, stream>>>(eIn, Bconv, dconv_b, Bhead,
                                         type_b, lam_b, dts, (float*)d_out);
}
